// MHAStateEncoder_49082886259301
// MI455X (gfx1250) — matrix operand order repaired and checked by reading, not compiled, not run on hardware
//
#include <hip/hip_runtime.h>
#include <hip/hip_bf16.h>

// MI455X (gfx1250) fused MHA state encoder.
// All four projections (Q,K,V,O) run on v_wmma_f32_16x16x32_bf16.
// One 256-thread / 8-wave32 workgroup handles 16 batches; K and V time-share
// one LDS buffer (attention weights live in registers across the barrier).

typedef __bf16 bf16_t;
typedef __attribute__((ext_vector_type(8)))  __bf16 v8bf;
typedef __attribute__((ext_vector_type(16))) __bf16 v16bf;
typedef __attribute__((ext_vector_type(8)))  float  v8f;

#define BATCH     16384
#define MAXN      20
#define DIM       64
#define DMODEL    128
#define NHEAD     8
#define DHEAD     16
#define SDIM      64
#define BPW       16                 // batches per workgroup
#define ROWS      (BPW * MAXN)       // 320 node rows per workgroup
#define NTHREADS  256

// ---- LDS layout (bytes); all offsets 32B-aligned ----
#define OFF_WQT   0                        // [128][192] bf16 = 49152
#define OFF_WKT   (OFF_WQT  + 49152)       // [128][64]  bf16 = 16384
#define OFF_WVT   (OFF_WKT  + 16384)       // [128][64]  bf16 = 16384
#define OFF_WOT   (OFF_WVT  + 16384)       // [64][128]  bf16 = 16384
#define OFF_NODES (OFF_WOT  + 16384)       // [320][64]  bf16 = 40960
#define OFF_KV    (OFF_NODES+ 40960)       // [320][128] bf16 = 81920 (K then V)
#define OFF_RAW   (OFF_KV   + 81920)       // [16][192]  bf16 = 6144 (aliased as ctx[16][128])
#define OFF_Q     (OFF_RAW  + 6144)        // [16][128]  bf16 = 4096
#define OFF_MASK  (OFF_Q    + 4096)        // [320]      f32  = 1280
#define SMEM_BYTES (OFF_MASK + 1280)       // 232704 bytes < 320KB/WGP

// A fragment (16x32 bf16, row-major [m][k]): lane m (0..15, dup 16..31),
// elems 0..7 = K[k0+half*8 .. +7], elems 8..15 = K[k0+16+half*8 .. +7].
__device__ __forceinline__ v16bf load_frag_a(const bf16_t* mat, int lda,
                                             int m0, int k0, int lane) {
    const int half = lane >> 4;
    const int m    = lane & 15;
    const bf16_t* p = mat + (m0 + m) * lda + k0 + half * 8;
    v8bf lo = *(const v8bf*)(p);        // 16B -> ds_load_b128
    v8bf hi = *(const v8bf*)(p + 16);   // 16B -> ds_load_b128
    return __builtin_shufflevector(lo, hi, 0,1,2,3,4,5,6,7,8,9,10,11,12,13,14,15);
}

// B fragment (32x16 bf16) from an [n][k]-transposed weight tile: lane n,
// elems 0..7 = K[k0+half*8 .. +7], elems 8..15 = K[k0+16+half*8 .. +7].
__device__ __forceinline__ v16bf load_frag_b(const bf16_t* matT, int ldb,
                                             int n0, int k0, int lane) {
    const int half = lane >> 4;
    const int n    = lane & 15;
    const bf16_t* p = matT + (n0 + n) * ldb + k0 + half * 8;
    v8bf lo = *(const v8bf*)(p);
    v8bf hi = *(const v8bf*)(p + 16);
    return __builtin_shufflevector(lo, hi, 0,1,2,3,4,5,6,7,8,9,10,11,12,13,14,15);
}

__device__ __forceinline__ v8f wmma_bf16(v16bf a, v16bf b, v8f c) {
    return __builtin_amdgcn_wmma_f32_16x16x32_bf16(false, a, false, b,
                                                   (short)0, c, false, false);
}

__global__ __launch_bounds__(NTHREADS)
void mha_state_encoder_kernel(const float* __restrict__ ne,
                              const int*   __restrict__ sidx,
                              const int*   __restrict__ eidx,
                              const int*   __restrict__ pidx,
                              const float* __restrict__ Wq, const float* __restrict__ bq,
                              const float* __restrict__ Wk, const float* __restrict__ bk,
                              const float* __restrict__ Wv, const float* __restrict__ bv,
                              const float* __restrict__ Wo, const float* __restrict__ bo,
                              float* __restrict__ out) {
    extern __shared__ char smem[];
    bf16_t* sWqT   = (bf16_t*)(smem + OFF_WQT);    // [n:128][k:192]
    bf16_t* sWkT   = (bf16_t*)(smem + OFF_WKT);    // [n:128][k:64]
    bf16_t* sWvT   = (bf16_t*)(smem + OFF_WVT);    // [n:128][k:64]
    bf16_t* sWoT   = (bf16_t*)(smem + OFF_WOT);    // [n:64][k:128]
    bf16_t* sNodes = (bf16_t*)(smem + OFF_NODES);  // [row:320][d:64]
    bf16_t* sKV    = (bf16_t*)(smem + OFF_KV);     // [row:320][n:128]
    bf16_t* sRaw   = (bf16_t*)(smem + OFF_RAW);    // [b:16][k:192]
    bf16_t* sCtx   = (bf16_t*)(smem + OFF_RAW);    // alias: [b:16][n:128]
    bf16_t* sQ     = (bf16_t*)(smem + OFF_Q);      // [b:16][n:128]
    float*  sMask  = (float*) (smem + OFF_MASK);   // [row:320] pre-scaled -1e9

    const int tid  = threadIdx.x;
    const int lane = tid & 31;
    const int wave = tid >> 5;
    const int half = lane >> 4;
    const int nlow = lane & 15;
    const int b0   = blockIdx.x * BPW;    // first batch of this WG
    const int r0   = blockIdx.x * ROWS;   // first global node row

    // ---------------- Phase 1: stage everything into LDS ----------------
    // Weights, transposed to [N][K] so B-fragments are contiguous LDS reads.
    for (int i = tid; i < 192 * DMODEL; i += NTHREADS) {
        int d = i / DMODEL, n = i % DMODEL;
        sWqT[n * 192 + d] = (bf16_t)Wq[i];
    }
    for (int i = tid; i < DIM * DMODEL; i += NTHREADS) {
        int d = i / DMODEL, n = i % DMODEL;
        sWkT[n * DIM + d] = (bf16_t)Wk[i];
        sWvT[n * DIM + d] = (bf16_t)Wv[i];
    }
    for (int i = tid; i < DMODEL * SDIM; i += NTHREADS) {
        int d = i / SDIM, n = i % SDIM;
        sWoT[n * DMODEL + d] = (bf16_t)Wo[i];
    }
    // Masked node rows (bf16) + softmax mask (row sum of masked values == 0).
    for (int r = tid; r < ROWS; r += NTHREADS) {
        const int gr = r0 + r;
        const bool ok = pidx[gr] >= 0;
        const float* src = ne + (size_t)gr * DIM;
        float rs = 0.f;
        #pragma unroll
        for (int d = 0; d < DIM; d += 4) {
            float4 x = ok ? *(const float4*)(src + d) : make_float4(0.f, 0.f, 0.f, 0.f);
            rs += x.x + x.y + x.z + x.w;
            sNodes[r * DIM + d + 0] = (bf16_t)x.x;
            sNodes[r * DIM + d + 1] = (bf16_t)x.y;
            sNodes[r * DIM + d + 2] = (bf16_t)x.z;
            sNodes[r * DIM + d + 3] = (bf16_t)x.w;
        }
        sMask[r] = (rs == 0.f) ? -1.0e9f : 0.f;
    }
    // raw = [agg | start | end] (agg uses UNMASKED rows per the reference).
    for (int i = tid; i < BPW * DIM; i += NTHREADS) {
        const int b = i / DIM, d = i % DIM;
        const float* src = ne + (size_t)(r0 + b * MAXN) * DIM + d;
        float s = 0.f;
        #pragma unroll
        for (int n_ = 0; n_ < MAXN; ++n_) s += src[(size_t)n_ * DIM];
        sRaw[b * 192 + d]        = (bf16_t)s;
        sRaw[b * 192 + 64 + d]   = (bf16_t)ne[(size_t)sidx[b0 + b] * DIM + d];
        sRaw[b * 192 + 128 + d]  = (bf16_t)ne[(size_t)eidx[b0 + b] * DIM + d];
    }
    // gfx1250 prefetch of next block's node rows into L2 (global_prefetch_b8).
    if (blockIdx.x + 1 < gridDim.x)
        __builtin_prefetch(ne + (size_t)(r0 + ROWS) * DIM + tid * 64, 0, 1);
    __syncthreads();

    // ---------------- Phase 2: Q GEMM + K GEMM (WMMA) ----------------
    {   // Q: raw(16x192) @ Wq(192x128): wave w owns N-tile w, 6 K-steps.
        v8f acc = {};
        #pragma unroll
        for (int ks = 0; ks < 6; ++ks) {
            v16bf a = load_frag_a(sRaw, 192, 0, ks * 32, lane);
            v16bf b = load_frag_b(sWqT, 192, wave * 16, ks * 32, lane);
            acc = wmma_bf16(a, b, acc);
        }
        const int n = wave * 16 + nlow;
        const float bias = bq[n];
        #pragma unroll
        for (int r = 0; r < 8; ++r) {
            const int m = half * 8 + r;   // D layout: vgpr r -> row r / r+8
            sQ[m * DMODEL + n] = (bf16_t)(acc[r] + bias);
        }
    }
    // K: nodes(320x64) @ Wk(64x128): 20 M-tiles x 8 N-tiles over 8 waves.
    for (int j = wave; j < (ROWS / 16) * 8; j += 8) {
        const int mt = j >> 3, nt = j & 7;
        v8f acc = {};
        #pragma unroll
        for (int ks = 0; ks < 2; ++ks) {
            v16bf a = load_frag_a(sNodes, DIM, mt * 16, ks * 32, lane);
            v16bf b = load_frag_b(sWkT, DIM, nt * 16, ks * 32, lane);
            acc = wmma_bf16(a, b, acc);
        }
        const int n = nt * 16 + nlow;
        const float bias = bk[n];
        #pragma unroll
        for (int r = 0; r < 8; ++r) {
            const int m = mt * 16 + half * 8 + r;
            sKV[m * DMODEL + n] = (bf16_t)(acc[r] + bias);
        }
    }
    __syncthreads();

    // ---------------- Phase 3: scores + softmax (VALU, 1 thread / (b,h)) ----
    float attn[MAXN];
    float inv_sum = 0.f;
    const int bh_b = tid >> 3, bh_h = tid & 7;
    if (tid < BPW * NHEAD) {
        float qv[DHEAD];
        #pragma unroll
        for (int i = 0; i < DHEAD; ++i)
            qv[i] = (float)sQ[bh_b * DMODEL + bh_h * DHEAD + i];
        float smax = -1.0e30f;
        #pragma unroll
        for (int n_ = 0; n_ < MAXN; ++n_) {
            const bf16_t* kr = sKV + (bh_b * MAXN + n_) * DMODEL + bh_h * DHEAD;
            float acc = 0.f;
            #pragma unroll
            for (int i = 0; i < DHEAD; ++i) acc += qv[i] * (float)kr[i];
            const float s = acc * 0.25f + sMask[bh_b * MAXN + n_];  // 1/sqrt(16)
            attn[n_] = s;
            smax = fmaxf(smax, s);
        }
        float ssum = 0.f;
        #pragma unroll
        for (int n_ = 0; n_ < MAXN; ++n_) {
            attn[n_] = __expf(attn[n_] - smax);
            ssum += attn[n_];
        }
        inv_sum = 1.f / ssum;
    }
    __syncthreads();   // everyone done reading K before V overwrites sKV

    // ---------------- Phase 4: V GEMM into the same buffer ----------------
    for (int j = wave; j < (ROWS / 16) * 8; j += 8) {
        const int mt = j >> 3, nt = j & 7;
        v8f acc = {};
        #pragma unroll
        for (int ks = 0; ks < 2; ++ks) {
            v16bf a = load_frag_a(sNodes, DIM, mt * 16, ks * 32, lane);
            v16bf b = load_frag_b(sWvT, DIM, nt * 16, ks * 32, lane);
            acc = wmma_bf16(a, b, acc);
        }
        const int n = nt * 16 + nlow;
        const float bias = bv[n];
        #pragma unroll
        for (int r = 0; r < 8; ++r) {
            const int m = mt * 16 + half * 8 + r;
            sKV[m * DMODEL + n] = (bf16_t)(acc[r] + bias);
        }
    }
    __syncthreads();

    // ---------------- Phase 5: ctx = attn . V (VALU) ----------------
    if (tid < BPW * NHEAD) {
        float ctx[DHEAD] = {};
        #pragma unroll
        for (int n_ = 0; n_ < MAXN; ++n_) {
            const float w = attn[n_] * inv_sum;
            const bf16_t* vr = sKV + (bh_b * MAXN + n_) * DMODEL + bh_h * DHEAD;
            #pragma unroll
            for (int i = 0; i < DHEAD; ++i) ctx[i] += w * (float)vr[i];
        }
        #pragma unroll
        for (int i = 0; i < DHEAD; ++i)
            sCtx[bh_b * DMODEL + bh_h * DHEAD + i] = (bf16_t)ctx[i];
    }
    __syncthreads();

    // ---------------- Phase 6: out = ctx(16x128) @ Wo(128x64) + bo --------
    if (wave < 4) {   // wave-uniform branch; EXEC stays all-ones for WMMA
        v8f acc = {};
        #pragma unroll
        for (int ks = 0; ks < 4; ++ks) {
            v16bf a = load_frag_a(sCtx, DMODEL, 0, ks * 32, lane);
            v16bf b = load_frag_b(sWoT, DMODEL, wave * 16, ks * 32, lane);
            acc = wmma_bf16(a, b, acc);
        }
        const int n = wave * 16 + nlow;
        const float bias = bo[n];
        #pragma unroll
        for (int r = 0; r < 8; ++r) {
            const int m = half * 8 + r;   // batch within this WG
            out[(size_t)(b0 + m) * SDIM + n] = acc[r] + bias;
        }
    }
}

extern "C" void kernel_launch(void* const* d_in, const int* in_sizes, int n_in,
                              void* d_out, int out_size, void* d_ws, size_t ws_size,
                              hipStream_t stream) {
    const float* node_embed = (const float*)d_in[0];
    const int*   start_idx  = (const int*)  d_in[1];
    const int*   end_idx    = (const int*)  d_in[2];
    // d_in[3] = seg_ids: implicit (row / MAX_NODES), unused on device.
    const int*   pad_idx    = (const int*)  d_in[4];
    const float* Wq = (const float*)d_in[5];
    const float* bq = (const float*)d_in[6];
    const float* Wk = (const float*)d_in[7];
    const float* bk = (const float*)d_in[8];
    const float* Wv = (const float*)d_in[9];
    const float* bv = (const float*)d_in[10];
    const float* Wo = (const float*)d_in[11];
    const float* bo = (const float*)d_in[12];
    float* out = (float*)d_out;

    dim3 grid(BATCH / BPW);   // 1024 workgroups
    dim3 block(NTHREADS);     // 8 wave32s
    mha_state_encoder_kernel<<<grid, block, SMEM_BYTES, stream>>>(
        node_embed, start_idx, end_idx, pad_idx,
        Wq, bq, Wk, bk, Wv, bv, Wo, bo, out);
}
